// DiT_Llama_60687887892845
// MI455X (gfx1250) — matrix operand order repaired and checked by reading, not compiled, not run on hardware
//
#include <hip/hip_runtime.h>
#include <hip/hip_bf16.h>
#include <stdint.h>
#include <stddef.h>

// ---------------- types ----------------
typedef _Float16 half_t;
typedef __attribute__((ext_vector_type(16))) _Float16 v16h;
typedef __attribute__((ext_vector_type(8)))  _Float16 h8;
typedef __attribute__((ext_vector_type(8)))  float    v8f;

#define BB 32
#define SS 256
#define DIMM 1024
#define NHH 16
#define HDD 64
#define HIDD 2816

static __device__ __forceinline__ float siluf(float v) {
    return v * (1.0f / (1.0f + expf(-v)));
}

static __device__ __forceinline__ v8f wmma16(v16h a, v16h b, v8f c) {
    return __builtin_amdgcn_wmma_f32_16x16x32_f16(false, a, false, b, (short)0, c, false, false);
}

// A-operand fragment (16x32, row-major source): lane<16 -> K {0..7,16..23}, lane>=16 -> K {8..15,24..31}
static __device__ __forceinline__ v16h load_fragA(const half_t* base, int stride, int lane) {
    int row = lane & 15;
    int kb  = (lane < 16) ? 0 : 8;
    const half_t* p = base + row * stride + kb;
    h8 lo = *(const h8*)(p);
    h8 hi = *(const h8*)(p + 16);
    v16h f;
#pragma unroll
    for (int i = 0; i < 8; ++i) { f[i] = lo[i]; f[i + 8] = hi[i]; }
    return f;
}

// B-operand fragment (32x16) from W stored [N,K] row-major: lane<16 -> K {0..7,16..23}, lane>=16 -> K {8..15,24..31}
static __device__ __forceinline__ v16h load_fragB(const half_t* base, int stride, int lane) {
    int col = lane & 15;
    int kb  = (lane < 16) ? 0 : 8;
    const half_t* p = base + col * stride + kb;
    h8 lo = *(const h8*)(p);
    h8 hi = *(const h8*)(p + 16);
    v16h f;
#pragma unroll
    for (int i = 0; i < 8; ++i) { f[i] = lo[i]; f[i + 8] = hi[i]; }
    return f;
}

// ---------------- generic WMMA GEMM: D = A[M,K] * W[N,K]^T ----------------
struct EpiParams {
    float*  outF;
    half_t* outH;
    const float* bias;
    const float* modp;   // EPI2: gate source
    int modStride;
    int gateOff;
    int S, NH, HD;       // EPI3
};

template <int EPI>
__global__ __launch_bounds__(256) void k_gemm(const half_t* __restrict__ A,
                                              const half_t* __restrict__ W,
                                              int M, int N, int K, EpiParams ep) {
    const int tid  = threadIdx.x;
    const int lane = tid & 31;
    const int wid  = tid >> 5;
    const int row0 = blockIdx.y * 128;
    const int col0 = blockIdx.x * 128;
    constexpr int STR = 40;
    __shared__ __align__(16) half_t As[2][128 * STR];
    __shared__ __align__(16) half_t Ws[2][128 * STR];

    v8f acc[2][4] = {};
    const int waveM = wid & 3;   // 4 waves along M (32 rows each)
    const int waveN = wid >> 2;  // 2 waves along N (64 cols each)
    const int nk = K >> 5;

    auto loadStage = [&](int buf, int ks) {
        int k0 = ks * 32;
#pragma unroll
        for (int j = 0; j < 2; ++j) {
            int c   = tid + j * 256;
            int r   = c >> 2;
            int seg = c & 3;
            int gr  = row0 + r;
            h8 va = {};
            if (gr < M) va = *(const h8*)(A + (size_t)gr * K + k0 + seg * 8);
            *(h8*)(&As[buf][r * STR + seg * 8]) = va;
            int gc = col0 + r;
            h8 vw = {};
            if (gc < N) vw = *(const h8*)(W + (size_t)gc * K + k0 + seg * 8);
            *(h8*)(&Ws[buf][r * STR + seg * 8]) = vw;
        }
    };

    loadStage(0, 0);
    __syncthreads();
    for (int ks = 0; ks < nk; ++ks) {
        int cur = ks & 1;
        if (ks + 1 < nk) loadStage(cur ^ 1, ks + 1);
        if (ks + 2 < nk) {
            int prRow = row0 + (tid >> 1);
            if (prRow < M)
                __builtin_prefetch((const void*)(A + (size_t)prRow * K + (size_t)(ks + 2) * 32 + (tid & 1) * 16), 0, 1);
        }
        v16h af[2], bf[4];
#pragma unroll
        for (int sm = 0; sm < 2; ++sm)
            af[sm] = load_fragA(&As[cur][(waveM * 32 + sm * 16) * STR], STR, lane);
#pragma unroll
        for (int sn = 0; sn < 4; ++sn)
            bf[sn] = load_fragB(&Ws[cur][(waveN * 64 + sn * 16) * STR], STR, lane);
#pragma unroll
        for (int sm = 0; sm < 2; ++sm)
#pragma unroll
            for (int sn = 0; sn < 4; ++sn)
                acc[sm][sn] = wmma16(af[sm], bf[sn], acc[sm][sn]);
        __syncthreads();
    }

#pragma unroll
    for (int sm = 0; sm < 2; ++sm)
#pragma unroll
        for (int sn = 0; sn < 4; ++sn)
#pragma unroll
            for (int r = 0; r < 8; ++r) {
                int gm = row0 + waveM * 32 + sm * 16 + r + ((lane < 16) ? 0 : 8);
                int gn = col0 + waveN * 64 + sn * 16 + (lane & 15);
                if (gm < M && gn < N) {
                    float v = acc[sm][sn][r];
                    if (EPI == 0) {
                        if (ep.bias) v += ep.bias[gn];
                        ep.outF[(size_t)gm * N + gn] = v;
                    } else if (EPI == 1) {
                        v += ep.bias[gn];
                        ep.outH[(size_t)gm * N + gn] = (half_t)v;
                    } else if (EPI == 2) {
                        float gate = ep.modp[(size_t)(gm / ep.S) * ep.modStride + ep.gateOff + gn];
                        if (ep.bias) v += ep.bias[gn];
                        ep.outF[(size_t)gm * N + gn] += gate * v;
                    } else { // EPI==3: V scatter -> [b,h,d,s]
                        int b = gm / ep.S, s = gm % ep.S;
                        int h = gn / ep.HD, d = gn % ep.HD;
                        ep.outH[(((size_t)(b * ep.NH + h) * ep.HD + d) * ep.S) + s] = (half_t)v;
                    }
                }
            }
}

// ---------------- conv2 implicit GEMM: M=32768 (b,y,x), N=512, K=25*512 ----------------
__global__ __launch_bounds__(256) void k_conv2(const half_t* __restrict__ inPad,
                                               const half_t* __restrict__ Wc2,
                                               const float* __restrict__ bias,
                                               float* __restrict__ h2) {
    const int tid  = threadIdx.x;
    const int lane = tid & 31;
    const int wid  = tid >> 5;
    const int row0 = blockIdx.y * 128;
    const int col0 = blockIdx.x * 128;
    constexpr int STR = 40;
    __shared__ __align__(16) half_t As[2][128 * STR];
    __shared__ __align__(16) half_t Ws[2][128 * STR];
    v8f acc[2][4] = {};
    const int waveM = wid & 3;
    const int waveN = wid >> 2;
    const int nk = 12800 / 32;

    auto loadStage = [&](int buf, int ks) {
        int k0  = ks * 32;
        int tap = k0 >> 9;
        int ci0 = k0 & 511;
        int kh = tap / 5, kw = tap % 5;
#pragma unroll
        for (int j = 0; j < 2; ++j) {
            int c   = tid + j * 256;
            int r   = c >> 2;
            int seg = c & 3;
            int m   = row0 + r;
            int b = m >> 10, rem = m & 1023, y = rem >> 5, x = rem & 31;
            h8 va = *(const h8*)(inPad + (size_t)((b * 36 + y + kh) * 36 + (x + kw)) * 512 + ci0 + seg * 8);
            *(h8*)(&As[buf][r * STR + seg * 8]) = va;
            int gc = col0 + r;
            h8 vw = *(const h8*)(Wc2 + (size_t)gc * 12800 + k0 + seg * 8);
            *(h8*)(&Ws[buf][r * STR + seg * 8]) = vw;
        }
    };

    loadStage(0, 0);
    __syncthreads();
    for (int ks = 0; ks < nk; ++ks) {
        int cur = ks & 1;
        if (ks + 1 < nk) loadStage(cur ^ 1, ks + 1);
        v16h af[2], bf[4];
#pragma unroll
        for (int sm = 0; sm < 2; ++sm)
            af[sm] = load_fragA(&As[cur][(waveM * 32 + sm * 16) * STR], STR, lane);
#pragma unroll
        for (int sn = 0; sn < 4; ++sn)
            bf[sn] = load_fragB(&Ws[cur][(waveN * 64 + sn * 16) * STR], STR, lane);
#pragma unroll
        for (int sm = 0; sm < 2; ++sm)
#pragma unroll
            for (int sn = 0; sn < 4; ++sn)
                acc[sm][sn] = wmma16(af[sm], bf[sn], acc[sm][sn]);
        __syncthreads();
    }
#pragma unroll
    for (int sm = 0; sm < 2; ++sm)
#pragma unroll
        for (int sn = 0; sn < 4; ++sn)
#pragma unroll
            for (int r = 0; r < 8; ++r) {
                int gm = row0 + waveM * 32 + sm * 16 + r + ((lane < 16) ? 0 : 8);
                int gn = col0 + waveN * 64 + sn * 16 + (lane & 15);
                int b = gm >> 10, rem = gm & 1023;
                float v = siluf(acc[sm][sn][r] + bias[gn]);
                h2[((size_t)(b * 512 + gn) << 10) + rem] = v;
            }
}

// ---------------- attention: one block = (b, head, 32 q rows) ----------------
__global__ __launch_bounds__(256) void k_attn(const half_t* __restrict__ Qh,
                                              const half_t* __restrict__ Kh,
                                              const half_t* __restrict__ Vt,
                                              half_t* __restrict__ Obuf) {
    const int tid  = threadIdx.x;
    const int lane = tid & 31;
    const int wid  = tid >> 5;
    int blk = blockIdx.x;
    int qb = blk & 7;
    int hh = (blk >> 3) & 15;
    int bb = blk >> 7;
    const half_t* Q  = Qh + ((size_t)(bb * NHH + hh) * SS + qb * 32) * HDD;
    const half_t* Kp = Kh + ((size_t)(bb * NHH + hh) * SS) * HDD;
    const half_t* Vp = Vt + ((size_t)(bb * NHH + hh) * HDD) * SS;

    __shared__ float sc[32 * 264];
    __shared__ __align__(16) half_t pr[32 * 264];

    const int waveM = wid & 1;        // 2 q-tiles of 16
    const int nb0   = (wid >> 1) * 4; // 4 key-tiles of 16 per wave
    v8f acc[4] = {};
#pragma unroll
    for (int kk = 0; kk < 2; ++kk) {
        v16h a = load_fragA(Q + (waveM * 16) * HDD + kk * 32, HDD, lane);
#pragma unroll
        for (int t = 0; t < 4; ++t) {
            v16h bfr = load_fragB(Kp + ((nb0 + t) * 16) * HDD + kk * 32, HDD, lane);
            acc[t] = wmma16(a, bfr, acc[t]);
        }
    }
#pragma unroll
    for (int t = 0; t < 4; ++t)
#pragma unroll
        for (int r = 0; r < 8; ++r) {
            int qr = waveM * 16 + r + ((lane < 16) ? 0 : 8);
            int kc = (nb0 + t) * 16 + (lane & 15);
            sc[qr * 264 + kc] = acc[t][r] * 0.125f;
        }
    __syncthreads();

    if (tid < 32) {
        float mx = -1e30f;
        for (int c = 0; c < 256; ++c) mx = fmaxf(mx, sc[tid * 264 + c]);
        float sum = 0.f;
        for (int c = 0; c < 256; ++c) { float e = expf(sc[tid * 264 + c] - mx); sc[tid * 264 + c] = e; sum += e; }
        float inv = 1.f / sum;
        for (int c = 0; c < 256; ++c) pr[tid * 264 + c] = (half_t)(sc[tid * 264 + c] * inv);
    }
    __syncthreads();

    const int dcol = (wid >> 1) * 16; // 4 d-tiles of 16 cover HD=64
    v8f oacc = {};
#pragma unroll
    for (int kk = 0; kk < 8; ++kk) {
        v16h a   = load_fragA(pr + (waveM * 16) * 264 + kk * 32, 264, lane);
        v16h bfr = load_fragB(Vp + (size_t)dcol * SS + kk * 32, SS, lane);
        oacc = wmma16(a, bfr, oacc);
    }
#pragma unroll
    for (int r = 0; r < 8; ++r) {
        int srow = qb * 32 + waveM * 16 + r + ((lane < 16) ? 0 : 8);
        int d    = dcol + (lane & 15);
        Obuf[((size_t)(bb * SS + srow) * NHH + hh) * HDD + d] = (half_t)oacc[r];
    }
}

// ---------------- LayerNorm with adaLN modulation (pre or post) ----------------
__global__ __launch_bounds__(256) void k_ln_mod(const float* __restrict__ T, half_t* __restrict__ out,
                                                const float* __restrict__ modp, int modStride, int S,
                                                int preSh, int preSc, int postSh, int postSc,
                                                const float* __restrict__ g, const float* __restrict__ bv,
                                                float eps) {
    int m = blockIdx.x, tid = threadIdx.x;
    int b = m / S;
    __shared__ float u[1024];
    __shared__ float red[256];
    const float* mrow = modp ? (modp + (size_t)b * modStride) : nullptr;
    float ps = 0.f;
#pragma unroll
    for (int j = 0; j < 4; ++j) {
        int d = tid + j * 256;
        float v = T[(size_t)m * 1024 + d];
        if (preSh >= 0) v = v * (1.f + mrow[preSc + d]) + mrow[preSh + d];
        u[d] = v;
        ps += v;
    }
    red[tid] = ps; __syncthreads();
    for (int s2 = 128; s2 > 0; s2 >>= 1) { if (tid < s2) red[tid] += red[tid + s2]; __syncthreads(); }
    float mean = red[0] * (1.f / 1024.f);
    __syncthreads();
    float pv = 0.f;
#pragma unroll
    for (int j = 0; j < 4; ++j) { int d = tid + j * 256; float dv = u[d] - mean; pv += dv * dv; }
    red[tid] = pv; __syncthreads();
    for (int s2 = 128; s2 > 0; s2 >>= 1) { if (tid < s2) red[tid] += red[tid + s2]; __syncthreads(); }
    float rstd = rsqrtf(red[0] * (1.f / 1024.f) + eps);
#pragma unroll
    for (int j = 0; j < 4; ++j) {
        int d = tid + j * 256;
        float v = (u[d] - mean) * rstd * g[d] + bv[d];
        if (postSh >= 0) v = v * (1.f + mrow[postSc + d]) + mrow[postSh + d];
        out[(size_t)m * 1024 + d] = (half_t)v;
    }
}

// ---------------- QK-norm + RoPE, writes Q/K as [b,h,s,d] f16 ----------------
__global__ __launch_bounds__(256) void k_qkrope(const float* __restrict__ qbuf, const float* __restrict__ kbuf,
                                                const float* __restrict__ qg, const float* __restrict__ qb2,
                                                const float* __restrict__ kg, const float* __restrict__ kb2,
                                                half_t* __restrict__ Qh, half_t* __restrict__ Kh) {
    int m = blockIdx.x, tid = threadIdx.x;
    int b = m >> 8, s = m & 255;
    __shared__ float u[1024];
    __shared__ float red[256];
    for (int which = 0; which < 2; ++which) {
        const float* src = which ? kbuf : qbuf;
        const float* g   = which ? kg : qg;
        const float* bv  = which ? kb2 : qb2;
        half_t* out      = which ? Kh : Qh;
        float ps = 0.f;
#pragma unroll
        for (int j = 0; j < 4; ++j) { int d = tid + j * 256; float v = src[(size_t)m * 1024 + d]; u[d] = v; ps += v; }
        red[tid] = ps; __syncthreads();
        for (int s2 = 128; s2 > 0; s2 >>= 1) { if (tid < s2) red[tid] += red[tid + s2]; __syncthreads(); }
        float mean = red[0] * (1.f / 1024.f);
        __syncthreads();
        float pv = 0.f;
#pragma unroll
        for (int j = 0; j < 4; ++j) { int d = tid + j * 256; float dv = u[d] - mean; pv += dv * dv; }
        red[tid] = pv; __syncthreads();
        for (int s2 = 128; s2 > 0; s2 >>= 1) { if (tid < s2) red[tid] += red[tid + s2]; __syncthreads(); }
        float rstd = rsqrtf(red[0] * (1.f / 1024.f) + 1e-5f);
        __syncthreads();
#pragma unroll
        for (int j = 0; j < 4; ++j) { int d = tid + j * 256; u[d] = (u[d] - mean) * rstd * g[d] + bv[d]; }
        __syncthreads();
        for (int p = tid; p < 512; p += 256) {
            int h = p >> 5, jj = p & 31;
            int d0 = h * 64 + 2 * jj;
            float fr  = powf(10000.f, -((float)(2 * jj)) / 64.f);
            float ang = (float)s * fr;
            float cs = cosf(ang), sn = sinf(ang);
            float qa = u[d0], qbv = u[d0 + 1];
            size_t base = (((size_t)(b * NHH + h) * SS) + s) * HDD;
            out[base + 2 * jj]     = (half_t)(qa * cs - qbv * sn);
            out[base + 2 * jj + 1] = (half_t)(qa * sn + qbv * cs);
        }
        __syncthreads();
    }
}

// ---------------- small elementwise / embedding kernels ----------------
__global__ void k_cast(const float* __restrict__ s, half_t* __restrict__ d, size_t n) {
    size_t i = (size_t)blockIdx.x * blockDim.x + threadIdx.x;
    if (i < n) d[i] = (half_t)s[i];
}

__global__ void k_conv1(const float* __restrict__ x, const float* __restrict__ w,
                        const float* __restrict__ bias, float* __restrict__ out) {
    size_t idx = (size_t)blockIdx.x * blockDim.x + threadIdx.x;
    if (idx >= (size_t)BB * 512 * 1024) return;
    int b = (int)(idx >> 19);
    int rem = (int)(idx & ((1u << 19) - 1));
    int co = rem >> 10;
    int pos = rem & 1023;
    int y = pos >> 5, xx = pos & 31;
    float acc = bias[co];
    for (int ci = 0; ci < 3; ++ci)
        for (int kh = 0; kh < 5; ++kh) {
            int yy = y + kh - 2;
            if (yy < 0 || yy >= 32) continue;
            for (int kw = 0; kw < 5; ++kw) {
                int xw = xx + kw - 2;
                if (xw < 0 || xw >= 32) continue;
                acc += x[((size_t)(b * 3 + ci) << 10) + yy * 32 + xw] * w[((co * 3 + ci) * 5 + kh) * 5 + kw];
            }
        }
    out[idx] = siluf(acc);
}

__global__ __launch_bounds__(256) void k_gn_stats(const float* __restrict__ src, float* __restrict__ mean,
                                                  float* __restrict__ rstd) {
    int bg = blockIdx.x;             // b*32 + group
    int b = bg >> 5, g = bg & 31;
    int tid = threadIdx.x;
    __shared__ float s1[256], s2[256];
    float a = 0.f, q = 0.f;
    for (int i = tid; i < 16384; i += 256) {
        int c = g * 16 + (i >> 10);
        int pos = i & 1023;
        float v = src[((size_t)(b * 512 + c) << 10) + pos];
        a += v; q += v * v;
    }
    s1[tid] = a; s2[tid] = q; __syncthreads();
    for (int st = 128; st > 0; st >>= 1) { if (tid < st) { s1[tid] += s1[tid + st]; s2[tid] += s2[tid + st]; } __syncthreads(); }
    if (tid == 0) {
        float m = s1[0] / 16384.f;
        float var = s2[0] / 16384.f - m * m;
        mean[bg] = m;
        rstd[bg] = rsqrtf(var + 1e-5f);
    }
}

__global__ __launch_bounds__(256) void k_gn1_pad(const float* __restrict__ h1, const float* __restrict__ mean,
                                                 const float* __restrict__ rstd, const float* __restrict__ g,
                                                 const float* __restrict__ bv, half_t* __restrict__ inPad) {
    int blk = blockIdx.x;            // b*36*36 + yy*36 + xx
    int b = blk / 1296;
    int rem = blk % 1296;
    int yy = rem / 36, xx = rem % 36;
    int y = yy - 2, x = xx - 2;
    bool inb = (y >= 0 && y < 32 && x >= 0 && x < 32);
    for (int c = threadIdx.x; c < 512; c += 256) {
        float v = 0.f;
        if (inb) {
            float raw = h1[((size_t)(b * 512 + c) << 10) + y * 32 + x];
            int grp = c >> 4;
            v = (raw - mean[b * 32 + grp]) * rstd[b * 32 + grp] * g[c] + bv[c];
        }
        inPad[((size_t)(b * 36 + yy) * 36 + xx) * 512 + c] = (half_t)v;
    }
}

__global__ void k_w2repack(const float* __restrict__ w2, half_t* __restrict__ Wc2) {
    size_t idx = (size_t)blockIdx.x * blockDim.x + threadIdx.x;
    if (idx >= (size_t)512 * 12800) return;
    int co = (int)(idx / 12800);
    int k = (int)(idx % 12800);
    int tap = k >> 9, ci = k & 511;
    int kh = tap / 5, kw = tap % 5;
    Wc2[idx] = (half_t)w2[(((size_t)co * 512 + ci) * 5 + kh) * 5 + kw];
}

__global__ __launch_bounds__(256) void k_gn2_patchify(const float* __restrict__ h2, const float* __restrict__ mean,
                                                      const float* __restrict__ rstd, const float* __restrict__ g,
                                                      const float* __restrict__ bv, half_t* __restrict__ tok_in) {
    int blk = blockIdx.x;            // b*256 + token
    int b = blk >> 8, tk = blk & 255;
    int gy = tk >> 4, gx = tk & 15;
    for (int f = threadIdx.x; f < 2048; f += 256) {
        int c = f >> 2;
        int py = (f >> 1) & 1, px = f & 1;
        float raw = h2[((size_t)(b * 512 + c) << 10) + (gy * 2 + py) * 32 + gx * 2 + px];
        int grp = c >> 4;
        float v = (raw - mean[b * 32 + grp]) * rstd[b * 32 + grp] * g[c] + bv[c];
        tok_in[(size_t)blk * 2048 + f] = (half_t)v;
    }
}

__global__ void k_temb1(const float* __restrict__ t, const float* __restrict__ w1,
                        const float* __restrict__ b1, float* __restrict__ out1) {
    int idx = blockIdx.x * blockDim.x + threadIdx.x;
    if (idx >= BB * 1024) return;
    int b = idx >> 10, o = idx & 1023;
    float tv = t[b];
    float acc = b1[o];
    const float lg = logf(10000.f);
    for (int i = 0; i < 256; ++i) {
        int ip = (i < 128) ? i : (i - 128);
        float fr = expf(-lg * (float)ip / 128.f);
        float arg = tv * fr;
        float v = (i < 128) ? cosf(arg) : sinf(arg);
        acc += v * w1[(size_t)o * 256 + i];
    }
    out1[idx] = siluf(acc);
}

__global__ void k_temb2(const float* __restrict__ in1, const float* __restrict__ w2,
                        const float* __restrict__ b2, const float* __restrict__ ytab,
                        const int* __restrict__ y, float* __restrict__ cbuf) {
    int idx = blockIdx.x * blockDim.x + threadIdx.x;
    if (idx >= BB * 1024) return;
    int b = idx >> 10, o = idx & 1023;
    float acc = b2[o];
    for (int i = 0; i < 1024; ++i) acc += in1[(size_t)b * 1024 + i] * w2[(size_t)o * 1024 + i];
    cbuf[idx] = acc + ytab[(size_t)y[b] * 1024 + o];
}

__global__ void k_silu_half(const float* __restrict__ s, half_t* __restrict__ d, int n) {
    int i = blockIdx.x * blockDim.x + threadIdx.x;
    if (i < n) d[i] = (half_t)siluf(s[i]);
}

__global__ void k_swiglu(const half_t* __restrict__ h1, half_t* __restrict__ mlpin) {
    size_t idx = (size_t)blockIdx.x * blockDim.x + threadIdx.x;
    if (idx >= (size_t)8192 * HIDD) return;
    size_t m = idx / HIDD;
    int j = (int)(idx % HIDD);
    float a = (float)h1[m * (2 * HIDD) + j];
    float g = (float)h1[m * (2 * HIDD) + HIDD + j];
    mlpin[idx] = (half_t)(siluf(a) * g);
}

__global__ void k_unpatchify(const float* __restrict__ xo, float* __restrict__ out) {
    int idx = blockIdx.x * blockDim.x + threadIdx.x;
    if (idx >= BB * 3 * 1024) return;
    int b = idx / 3072;
    int rem = idx % 3072;
    int c = rem >> 10;
    int hw = rem & 1023;
    int hh = hw >> 5, ww = hw & 31;
    int gy = hh >> 1, py = hh & 1;
    int gx = ww >> 1, px = ww & 1;
    int tk = gy * 16 + gx;
    int f = (py * 2 + px) * 3 + c;
    out[idx] = xo[(size_t)(b * 256 + tk) * 12 + f];
}

// ---------------- host ----------------
static void cast_w(const float* s, half_t* d, size_t n, hipStream_t st) {
    k_cast<<<(unsigned)((n + 255) / 256), 256, 0, st>>>(s, d, n);
}

static void gemm_f32(const half_t* A, const half_t* W, int M, int N, int K,
                     float* out, const float* bias, hipStream_t st) {
    EpiParams ep = {}; ep.outF = out; ep.bias = bias;
    dim3 g((N + 127) / 128, (M + 127) / 128);
    k_gemm<0><<<g, 256, 0, st>>>(A, W, M, N, K, ep);
}
static void gemm_f16(const half_t* A, const half_t* W, int M, int N, int K,
                     half_t* out, const float* bias, hipStream_t st) {
    EpiParams ep = {}; ep.outH = out; ep.bias = bias;
    dim3 g((N + 127) / 128, (M + 127) / 128);
    k_gemm<1><<<g, 256, 0, st>>>(A, W, M, N, K, ep);
}
static void gemm_gate(const half_t* A, const half_t* W, int M, int N, int K, float* T,
                      const float* bias, const float* modp, int modStride, int gateOff, hipStream_t st) {
    EpiParams ep = {}; ep.outF = T; ep.bias = bias; ep.modp = modp; ep.modStride = modStride;
    ep.gateOff = gateOff; ep.S = SS;
    dim3 g((N + 127) / 128, (M + 127) / 128);
    k_gemm<2><<<g, 256, 0, st>>>(A, W, M, N, K, ep);
}
static void gemm_vt(const half_t* A, const half_t* W, int M, int N, int K, half_t* Vt, hipStream_t st) {
    EpiParams ep = {}; ep.outH = Vt; ep.S = SS; ep.NH = NHH; ep.HD = HDD;
    dim3 g((N + 127) / 128, (M + 127) / 128);
    k_gemm<3><<<g, 256, 0, st>>>(A, W, M, N, K, ep);
}

extern "C" void kernel_launch(void* const* d_in, const int* in_sizes, int n_in,
                              void* d_out, int out_size, void* d_ws, size_t ws_size,
                              hipStream_t stream) {
    (void)in_sizes; (void)n_in; (void)out_size; (void)ws_size;
    const float* x       = (const float*)d_in[0];
    const float* t       = (const float*)d_in[1];
    const float* conv1_w = (const float*)d_in[2];
    const float* conv1_b = (const float*)d_in[3];
    const float* gn1_g   = (const float*)d_in[4];
    const float* gn1_b   = (const float*)d_in[5];
    const float* conv2_w = (const float*)d_in[6];
    const float* conv2_b = (const float*)d_in[7];
    const float* gn2_g   = (const float*)d_in[8];
    const float* gn2_b   = (const float*)d_in[9];
    const float* xemb_w  = (const float*)d_in[10];
    const float* xemb_b  = (const float*)d_in[11];
    const float* t_w1    = (const float*)d_in[12];
    const float* t_b1    = (const float*)d_in[13];
    const float* t_w2    = (const float*)d_in[14];
    const float* t_b2    = (const float*)d_in[15];
    const float* y_table = (const float*)d_in[16];
    const float* ada_w   = (const float*)d_in[17];
    const float* ada_b   = (const float*)d_in[18];
    const float* an_g    = (const float*)d_in[19];
    const float* an_b    = (const float*)d_in[20];
    const float* wq      = (const float*)d_in[21];
    const float* wk      = (const float*)d_in[22];
    const float* wv      = (const float*)d_in[23];
    const float* wo      = (const float*)d_in[24];
    const float* qn_g    = (const float*)d_in[25];
    const float* qn_b    = (const float*)d_in[26];
    const float* kn_g    = (const float*)d_in[27];
    const float* kn_b    = (const float*)d_in[28];
    const float* ffn_g   = (const float*)d_in[29];
    const float* ffn_b   = (const float*)d_in[30];
    const float* fc1_w   = (const float*)d_in[31];
    const float* fc1_b   = (const float*)d_in[32];
    const float* fc2_w   = (const float*)d_in[33];
    const float* fc2_b   = (const float*)d_in[34];
    const float* fn_g    = (const float*)d_in[35];
    const float* fn_b    = (const float*)d_in[36];
    const float* fin_w   = (const float*)d_in[37];
    const float* fin_b   = (const float*)d_in[38];
    const float* fada_w  = (const float*)d_in[39];
    const float* fada_b  = (const float*)d_in[40];
    const int*   yidx    = (const int*)d_in[41];
    float* outp = (float*)d_out;

    char* wp = (char*)d_ws;
    auto alloc = [&](size_t bytes) -> void* {
        void* r = (void*)wp;
        wp += (bytes + 255) & ~(size_t)255;
        return r;
    };
    const int M = BB * SS; // 8192

    half_t* Wx    = (half_t*)alloc((size_t)1024 * 2048 * 2);
    half_t* Wada  = (half_t*)alloc((size_t)4 * 6144 * 1024 * 2);
    half_t* Wq    = (half_t*)alloc((size_t)4 * 1024 * 1024 * 2);
    half_t* Wk    = (half_t*)alloc((size_t)4 * 1024 * 1024 * 2);
    half_t* Wv    = (half_t*)alloc((size_t)4 * 1024 * 1024 * 2);
    half_t* Wo    = (half_t*)alloc((size_t)4 * 1024 * 1024 * 2);
    half_t* Wfc1  = (half_t*)alloc((size_t)4 * 5632 * 1024 * 2);
    half_t* Wfc2  = (half_t*)alloc((size_t)4 * 1024 * 2816 * 2);
    half_t* Wfada = (half_t*)alloc((size_t)2048 * 1024 * 2);
    half_t* Wfin  = (half_t*)alloc((size_t)12 * 1024 * 2);
    half_t* Wc2   = (half_t*)alloc((size_t)512 * 12800 * 2);
    float*  h1c   = (float*)alloc((size_t)BB * 512 * 1024 * 4);  // conv1 out; later aliased as h2
    half_t* inPad = (half_t*)alloc((size_t)BB * 36 * 36 * 512 * 2);
    float*  mean1 = (float*)alloc(1024 * 4);
    float*  rstd1 = (float*)alloc(1024 * 4);
    float*  mean2 = (float*)alloc(1024 * 4);
    float*  rstd2 = (float*)alloc(1024 * 4);
    half_t* tok_in = (half_t*)alloc((size_t)M * 2048 * 2);
    float*  T      = (float*)alloc((size_t)M * 1024 * 4);
    float*  out1   = (float*)alloc((size_t)BB * 1024 * 4);
    float*  cbuf   = (float*)alloc((size_t)BB * 1024 * 4);
    half_t* cs     = (half_t*)alloc((size_t)BB * 1024 * 2);
    float*  modbuf = (float*)alloc((size_t)BB * 24576 * 4);
    float*  fmod   = (float*)alloc((size_t)BB * 2048 * 4);
    half_t* xn     = (half_t*)alloc((size_t)M * 1024 * 2);
    float*  qbuf   = (float*)alloc((size_t)M * 1024 * 4);
    float*  kbuf   = (float*)alloc((size_t)M * 1024 * 4);
    half_t* Qh     = (half_t*)alloc((size_t)M * 1024 * 2);
    half_t* Kh     = (half_t*)alloc((size_t)M * 1024 * 2);
    half_t* Vt     = (half_t*)alloc((size_t)M * 1024 * 2);
    half_t* Obuf   = (half_t*)alloc((size_t)M * 1024 * 2);
    half_t* h1f    = (half_t*)alloc((size_t)M * 5632 * 2);
    half_t* mlpin  = (half_t*)alloc((size_t)M * 2816 * 2);
    float*  xo     = (float*)alloc((size_t)M * 12 * 4);
    float*  h2     = h1c; // alias: h1c dead after gn1_pad

    // weight casts (fp32 -> f16, [N,K] row-major preserved)
    cast_w(xemb_w, Wx, (size_t)1024 * 2048, stream);
    cast_w(ada_w, Wada, (size_t)4 * 6144 * 1024, stream);
    cast_w(wq, Wq, (size_t)4 * 1024 * 1024, stream);
    cast_w(wk, Wk, (size_t)4 * 1024 * 1024, stream);
    cast_w(wv, Wv, (size_t)4 * 1024 * 1024, stream);
    cast_w(wo, Wo, (size_t)4 * 1024 * 1024, stream);
    cast_w(fc1_w, Wfc1, (size_t)4 * 5632 * 1024, stream);
    cast_w(fc2_w, Wfc2, (size_t)4 * 1024 * 2816, stream);
    cast_w(fada_w, Wfada, (size_t)2048 * 1024, stream);
    cast_w(fin_w, Wfin, (size_t)12 * 1024, stream);
    k_w2repack<<<(512 * 12800 + 255) / 256, 256, 0, stream>>>(conv2_w, Wc2);

    // timestep + class embedding
    k_temb1<<<(BB * 1024 + 255) / 256, 256, 0, stream>>>(t, t_w1, t_b1, out1);
    k_temb2<<<(BB * 1024 + 255) / 256, 256, 0, stream>>>(out1, t_w2, t_b2, y_table, yidx, cbuf);
    k_silu_half<<<(BB * 1024 + 255) / 256, 256, 0, stream>>>(cbuf, cs, BB * 1024);

    // conv stack
    k_conv1<<<(BB * 512 * 1024) / 256, 256, 0, stream>>>(x, conv1_w, conv1_b, h1c);
    k_gn_stats<<<BB * 32, 256, 0, stream>>>(h1c, mean1, rstd1);
    k_gn1_pad<<<BB * 36 * 36, 256, 0, stream>>>(h1c, mean1, rstd1, gn1_g, gn1_b, inPad);
    {
        dim3 g(512 / 128, 32768 / 128);
        k_conv2<<<g, 256, 0, stream>>>(inPad, Wc2, conv2_b, h2);
    }
    k_gn_stats<<<BB * 32, 256, 0, stream>>>(h2, mean2, rstd2);
    k_gn2_patchify<<<BB * 256, 256, 0, stream>>>(h2, mean2, rstd2, gn2_g, gn2_b, tok_in);

    // token embedding + adaLN tables
    gemm_f32(tok_in, Wx, M, 1024, 2048, T, xemb_b, stream);
    gemm_f32(cs, Wada, BB, 24576, 1024, modbuf, ada_b, stream);
    gemm_f32(cs, Wfada, BB, 2048, 1024, fmod, fada_b, stream);

    for (int i = 0; i < 4; ++i) {
        int mo = i * 6144;
        // attention branch
        k_ln_mod<<<M, 256, 0, stream>>>(T, xn, modbuf, 24576, SS, mo + 0, mo + 1024, -1, -1,
                                        an_g + i * 1024, an_b + i * 1024, 1e-5f);
        gemm_f32(xn, Wq + (size_t)i * 1024 * 1024, M, 1024, 1024, qbuf, nullptr, stream);
        gemm_f32(xn, Wk + (size_t)i * 1024 * 1024, M, 1024, 1024, kbuf, nullptr, stream);
        gemm_vt(xn, Wv + (size_t)i * 1024 * 1024, M, 1024, 1024, Vt, stream);
        k_qkrope<<<M, 256, 0, stream>>>(qbuf, kbuf, qn_g + i * 1024, qn_b + i * 1024,
                                        kn_g + i * 1024, kn_b + i * 1024, Qh, Kh);
        k_attn<<<BB * NHH * 8, 256, 0, stream>>>(Qh, Kh, Vt, Obuf);
        gemm_gate(Obuf, Wo + (size_t)i * 1024 * 1024, M, 1024, 1024, T, nullptr,
                  modbuf, 24576, mo + 2 * 1024, stream);
        // MLP branch
        k_ln_mod<<<M, 256, 0, stream>>>(T, xn, modbuf, 24576, SS, mo + 3 * 1024, mo + 4 * 1024, -1, -1,
                                        ffn_g + i * 1024, ffn_b + i * 1024, 1e-5f);
        gemm_f16(xn, Wfc1 + (size_t)i * 5632 * 1024, M, 5632, 1024, h1f, fc1_b + i * 5632, stream);
        k_swiglu<<<(unsigned)(((size_t)M * HIDD + 255) / 256), 256, 0, stream>>>(h1f, mlpin);
        gemm_gate(mlpin, Wfc2 + (size_t)i * 1024 * 2816, M, 1024, 2816, T, fc2_b + i * 1024,
                  modbuf, 24576, mo + 5 * 1024, stream);
    }

    // final layer
    k_ln_mod<<<M, 256, 0, stream>>>(T, xn, fmod, 2048, SS, -1, -1, 0, 1024, fn_g, fn_b, 1e-6f);
    gemm_f32(xn, Wfin, M, 12, 1024, xo, fin_b, stream);
    k_unpatchify<<<(BB * 3 * 1024 + 255) / 256, 256, 0, stream>>>(xo, outp);
}
